// EncoderLayer_1408749273738
// MI455X (gfx1250) — matrix operand order repaired and checked by reading, not compiled, not run on hardware
//
#include <hip/hip_runtime.h>

// ---------------------------------------------------------------------------
// CDNA5 WMMA types: D(f32 16x16) = A(bf16 16x32) * B(bf16 32x16) + C
// ---------------------------------------------------------------------------
typedef __attribute__((ext_vector_type(16))) __bf16 v16bf;
typedef __attribute__((ext_vector_type(8)))  float  v8f;

union FragBF16 { v16bf v; unsigned u[8]; };

__device__ __forceinline__ v8f zero_v8f() {
  v8f z;
#pragma unroll
  for (int i = 0; i < 8; ++i) z[i] = 0.0f;
  return z;
}

// native f32 -> bf16 (RNE hardware cvt), packed pair into one u32
__device__ __forceinline__ unsigned pack2_bf16(float a, float b) {
  union { __bf16 h[2]; unsigned u; } cv;
  cv.h[0] = (__bf16)a;
  cv.h[1] = (__bf16)b;
  return cv.u;
}

// A-operand (16x32 bf16) per-lane pair index within a 32-wide K window.
// lanes 0-15 K=0..15 (pair i -> K=2i), lanes 16-31 K=16..31 (same order as B).
__device__ __forceinline__ int a_pair_idx(int i, int h) {
  return h * 8 + i;
}
// B-operand (32x16 bf16): lanes 0-15 K=0..15 (pair i -> K=2i), lanes 16-31 K=16..31.
__device__ __forceinline__ int b_pair_idx(int i, int h) { return h * 8 + i; }

__device__ __forceinline__ v8f wmma_bf16(const FragBF16& a, const FragBF16& b, v8f c) {
  return __builtin_amdgcn_wmma_f32_16x16x32_bf16(false, a.v, false, b.v,
                                                 (short)0, c, false, false);
}

// ---------------------------------------------------------------------------
// CDNA5 async global->LDS copy (ASYNCcnt path), 16B per lane
// ---------------------------------------------------------------------------
__device__ __forceinline__ void async_copy_b128(unsigned lds_byte_off, const void* gaddr) {
  asm volatile("global_load_async_to_lds_b128 %0, %1, off"
               :: "v"(lds_byte_off), "v"(gaddr)
               : "memory");
}
__device__ __forceinline__ void wait_async0() {
  asm volatile("s_wait_asynccnt 0x0" ::: "memory");
}
__device__ __forceinline__ unsigned lds_off_u32(const void* p) {
  return (unsigned)(uintptr_t)p;  // low 32 bits of generic LDS pointer = group-segment offset
}

// ---------------------------------------------------------------------------
// fp32 -> packed bf16 pairs (one pass per matrix; halves GEMM operand traffic)
// ---------------------------------------------------------------------------
__global__ __launch_bounds__(256) void cvt_bf16_kernel(const float2* __restrict__ X,
                                                       unsigned* __restrict__ Y, int n2) {
  int i = blockIdx.x * 256 + threadIdx.x;
  if (i < n2) {
    float2 v = X[i];
    Y[i] = pack2_bf16(v.x, v.y);
  }
}

// ---------------------------------------------------------------------------
// GEMM: C[M,N] = A[M,K] * W[N,K]^T + bias (+ReLU). A2/W2 are packed-bf16
// (u32 = 2 elems, row pitch K/2). Block tile 128x128, K-step 32, 8 waves
// (4x2), each wave 32x64 (2x4 WMMA tiles -> 1.5 LDS b128 reads per WMMA).
// Double-buffered async-to-LDS staging (ASYNCcnt), one barrier per K-step.
// ---------------------------------------------------------------------------
#define GTM 128
#define GTN 128

__global__ __launch_bounds__(256) void gemm_bf16_kernel(
    const unsigned* __restrict__ A2, const unsigned* __restrict__ W2,
    const float* __restrict__ bias, float* __restrict__ C,
    int M, int N, int K, int relu) {
  __shared__ unsigned As[2][GTM][16];  // 2 x 8 KB
  __shared__ unsigned Ws[2][GTN][16];  // 2 x 8 KB

  const int tid  = threadIdx.x;
  const int lane = tid & 31;
  const int wave = tid >> 5;   // 0..7
  const int wm   = wave >> 1;  // 0..3 -> +32*wm rows
  const int wn   = wave & 1;   // 0..1 -> +64*wn cols
  const int m0 = blockIdx.x * GTM;
  const int n0 = blockIdx.y * GTN;
  const int h = lane >> 4, r16 = lane & 15;
  const int kpitch = K >> 1;  // u32 per row

  // per-thread staging: A: 8 u32 (two b128), W: 8 u32 (two b128)
  const int arow = tid >> 1, acp = (tid & 1) * 8;
  const unsigned aLds0 = lds_off_u32(&As[0][arow][acp]);
  const unsigned wLds0 = lds_off_u32(&Ws[0][arow][acp]);
  const unsigned aStride = GTM * 16 * 4;  // bytes between buffers
  const unsigned wStride = GTN * 16 * 4;
  const unsigned* aG = A2 + (size_t)(m0 + arow) * kpitch + acp;
  const unsigned* wG = W2 + (size_t)(n0 + arow) * kpitch + acp;

  v8f acc[2][4];
#pragma unroll
  for (int i = 0; i < 2; ++i)
#pragma unroll
    for (int j = 0; j < 4; ++j) acc[i][j] = zero_v8f();

  const int nk = K >> 5;  // K-steps of 32 elements (16 u32)

  // prologue: stage 0 into buffer 0
  async_copy_b128(aLds0, aG);
  async_copy_b128(aLds0 + 16, aG + 4);
  async_copy_b128(wLds0, wG);
  async_copy_b128(wLds0 + 16, wG + 4);

  for (int s = 0; s < nk; ++s) {
    const int buf = s & 1;
    wait_async0();       // this wave's copies into buf are done
    __syncthreads();     // everyone's copies done; prior reads of buf^1 done
    if (s + 1 < nk) {
      const unsigned* aN = aG + (size_t)(s + 1) * 16;
      const unsigned* wN = wG + (size_t)(s + 1) * 16;
      const unsigned ab = aLds0 + (buf ^ 1) * aStride;
      const unsigned wb = wLds0 + (buf ^ 1) * wStride;
      async_copy_b128(ab, aN);
      async_copy_b128(ab + 16, aN + 4);
      async_copy_b128(wb, wN);
      async_copy_b128(wb + 16, wN + 4);
    }

    FragBF16 af[2], bf[4];
#pragma unroll
    for (int i = 0; i < 8; ++i) {
      int ap = a_pair_idx(i, h);
      af[0].u[i] = As[buf][wm * 32 + r16][ap];
      af[1].u[i] = As[buf][wm * 32 + 16 + r16][ap];
      int bp = b_pair_idx(i, h);
#pragma unroll
      for (int j = 0; j < 4; ++j)
        bf[j].u[i] = Ws[buf][wn * 64 + j * 16 + r16][bp];
    }
#pragma unroll
    for (int i = 0; i < 2; ++i)
#pragma unroll
      for (int j = 0; j < 4; ++j) acc[i][j] = wmma_bf16(af[i], bf[j], acc[i][j]);
  }

#pragma unroll
  for (int i = 0; i < 2; ++i)
#pragma unroll
    for (int j = 0; j < 4; ++j) {
      int n = n0 + wn * 64 + j * 16 + r16;
      float bv = bias[n];
#pragma unroll
      for (int r = 0; r < 8; ++r) {
        int m = m0 + wm * 32 + i * 16 + r + 8 * h;  // C layout: M = r + 8*half
        float v = acc[i][j][r] + bv;
        if (relu) v = fmaxf(v, 0.0f);
        C[(size_t)m * N + n] = v;
      }
    }
}

// ---------------------------------------------------------------------------
// Attention: per (b, head, 64-query block). 4 waves x 16 queries. Online
// softmax over key chunks of KC=64; scores and PV on WMMA.
// ---------------------------------------------------------------------------
#define KC 64

__global__ __launch_bounds__(128) void attn_kernel(
    const float* __restrict__ Q, const float* __restrict__ Kmat,
    const float* __restrict__ V, const int* __restrict__ mask,
    float* __restrict__ O, int S, int H, int nheads) {
  __shared__ unsigned Ksh[KC][32];      // [key][d-pair]
  __shared__ unsigned Vsh[64][KC / 2];  // [d][key-pair] (transposed)
  __shared__ float    sc[4][16][KC];    // per-wave scores
  __shared__ float    srow_s[4][16], lrow_s[4][16];

  const int tid = threadIdx.x;
  const int lane = tid & 31;
  const int wave = tid >> 5;
  const int b  = blockIdx.x / nheads;
  const int hh = blockIdx.x % nheads;
  const int q0 = blockIdx.y * 64 + wave * 16;
  const int h = lane >> 4, r16 = lane & 15;
  const size_t baseBS = (size_t)b * S;
  const int hd = hh * 64;

  FragBF16 aq[2];
  {
    const float* qrow = Q + (baseBS + q0 + r16) * H + hd;
#pragma unroll
    for (int d = 0; d < 2; ++d)
#pragma unroll
      for (int i = 0; i < 8; ++i) {
        int kb = d * 32 + a_pair_idx(i, h) * 2;
        aq[d].u[i] = pack2_bf16(qrow[kb], qrow[kb + 1]);
      }
  }

  v8f acc[4];
#pragma unroll
  for (int t = 0; t < 4; ++t) acc[t] = zero_v8f();
  float mold = -1e30f, lold = 0.0f;  // lanes 0..15: one query row each

  for (int c = 0; c < S; c += KC) {
    __syncthreads();
#pragma unroll
    for (int p = tid; p < KC * 32; p += 128) {
      int row = p >> 5, cp = p & 31;
      const float* g = Kmat + (baseBS + c + row) * H + hd + cp * 2;
      Ksh[row][cp] = pack2_bf16(g[0], g[1]);
    }
#pragma unroll
    for (int p = tid; p < 64 * (KC / 2); p += 128) {
      int d = p & 63, kq = p >> 6;
      const float* g = V + (baseBS + c + kq * 2) * H + hd + d;
      Vsh[d][kq] = pack2_bf16(g[0], g[H]);
    }
    __syncthreads();

#pragma unroll
    for (int kt = 0; kt < 4; ++kt) {
      FragBF16 bk[2];
#pragma unroll
      for (int i = 0; i < 8; ++i) {
        int bp = b_pair_idx(i, h);
        bk[0].u[i] = Ksh[kt * 16 + r16][bp];
        bk[1].u[i] = Ksh[kt * 16 + r16][16 + bp];
      }
      v8f s = zero_v8f();
      s = wmma_bf16(aq[0], bk[0], s);
      s = wmma_bf16(aq[1], bk[1], s);
      int key = c + kt * 16 + r16;
      bool keep = mask[b * S + key] != 0;
#pragma unroll
      for (int r = 0; r < 8; ++r) {
        float x = s[r] * 0.125f;  // 1/sqrt(d_k)
        if (!keep) x = -1e9f;
        sc[wave][r + 8 * h][kt * 16 + r16] = x;
      }
    }
    __syncthreads();

    if (lane < 16) {
      float cm = -1e30f;
      for (int j = 0; j < KC; ++j) cm = fmaxf(cm, sc[wave][lane][j]);
      float mn = fmaxf(mold, cm);
      float ssc = __expf(mold - mn);
      float csum = 0.0f;
      for (int j = 0; j < KC; ++j) {
        float p = __expf(sc[wave][lane][j] - mn);
        sc[wave][lane][j] = p;
        csum += p;
      }
      lold = lold * ssc + csum;
      mold = mn;
      srow_s[wave][lane] = ssc;
      lrow_s[wave][lane] = lold;
    }
    __syncthreads();

#pragma unroll
    for (int r = 0; r < 8; ++r) {
      float sr = srow_s[wave][r + 8 * h];
#pragma unroll
      for (int t = 0; t < 4; ++t) acc[t][r] *= sr;
    }

#pragma unroll
    for (int ks = 0; ks < 2; ++ks) {
      FragBF16 ap;
#pragma unroll
      for (int i = 0; i < 8; ++i) {
        int kb = ks * 32 + a_pair_idx(i, h) * 2;
        ap.u[i] = pack2_bf16(sc[wave][r16][kb], sc[wave][r16][kb + 1]);
      }
#pragma unroll
      for (int dt = 0; dt < 4; ++dt) {
        FragBF16 bv;
#pragma unroll
        for (int i = 0; i < 8; ++i)
          bv.u[i] = Vsh[dt * 16 + r16][ks * 16 + b_pair_idx(i, h)];
        acc[dt] = wmma_bf16(ap, bv, acc[dt]);
      }
    }
  }
  __syncthreads();

#pragma unroll
  for (int dt = 0; dt < 4; ++dt)
#pragma unroll
    for (int r = 0; r < 8; ++r) {
      int m = r + 8 * h;
      float li = lrow_s[wave][m];
      O[(baseBS + q0 + m) * H + hd + dt * 16 + r16] = acc[dt][r] / li;
    }
}

// ---------------------------------------------------------------------------
// Y = g * (X + R - mean) / (std_unbiased + eps) + be  (torch semantics)
// ---------------------------------------------------------------------------
__global__ __launch_bounds__(256) void add_ln_kernel(
    const float* __restrict__ X, const float* __restrict__ R,
    const float* __restrict__ g, const float* __restrict__ be,
    float* __restrict__ Y, int H) {
  __shared__ float b1[256], b2[256];
  const int tid = threadIdx.x;
  const size_t row = (size_t)blockIdx.x * H;

  float s = 0.0f, ss = 0.0f;
  for (int i = tid; i < H; i += 256) {
    float v = X[row + i] + R[row + i];
    s += v;
    ss += v * v;
  }
  b1[tid] = s; b2[tid] = ss;
  __syncthreads();
  for (int off = 128; off > 0; off >>= 1) {
    if (tid < off) { b1[tid] += b1[tid + off]; b2[tid] += b2[tid + off]; }
    __syncthreads();
  }
  float mean = b1[0] / (float)H;
  float var = fmaxf((b2[0] - (float)H * mean * mean) / (float)(H - 1), 0.0f);
  float inv = 1.0f / (sqrtf(var) + 1e-6f);
  for (int i = tid; i < H; i += 256) {
    float v = X[row + i] + R[row + i];
    Y[row + i] = g[i] * (v - mean) * inv + be[i];
  }
}

// ---------------------------------------------------------------------------
// launch
// ---------------------------------------------------------------------------
extern "C" void kernel_launch(void* const* d_in, const int* in_sizes, int n_in,
                              void* d_out, int out_size, void* d_ws, size_t ws_size,
                              hipStream_t stream) {
  constexpr int B = 4, S = 1024, H = 1024, NH = 16;
  constexpr int M = B * S;  // 4096

  const float* x    = (const float*)d_in[0];
  const int*   mask = (const int*)  d_in[1];
  const float* wq = (const float*)d_in[2];  const float* bq = (const float*)d_in[3];
  const float* wk = (const float*)d_in[4];  const float* bk = (const float*)d_in[5];
  const float* wv = (const float*)d_in[6];  const float* bv = (const float*)d_in[7];
  const float* wo = (const float*)d_in[8];  const float* bo = (const float*)d_in[9];
  const float* w1 = (const float*)d_in[10]; const float* b1 = (const float*)d_in[11];
  const float* w2 = (const float*)d_in[12]; const float* b2 = (const float*)d_in[13];
  const float* g1 = (const float*)d_in[14]; const float* be1 = (const float*)d_in[15];
  const float* g2 = (const float*)d_in[16]; const float* be2 = (const float*)d_in[17];
  float* out = (float*)d_out;

  float* ws = (float*)d_ws;
  const size_t SLOT = (size_t)M * H;  // 4 Mi elems
  float* qb   = ws + 0 * SLOT;
  float* kb   = ws + 1 * SLOT;
  float* vb   = ws + 2 * SLOT;
  float* attn = ws + 3 * SLOT;
  float* t1   = ws + 4 * SLOT;
  unsigned* abuf = (unsigned*)(ws + 5 * SLOT);  // packed-bf16 activations (<= 4 Mi u32)
  unsigned* wbuf = (unsigned*)(ws + 9 * SLOT);  // packed-bf16 weights (<= 1 Mi u32)
  float* x1  = attn;  // reuse once attention is consumed
  float* ff1 = qb;    // reuse q+k slots (8 Mi floats)

  dim3 blk(256);
  auto cvt = [&](const float* src, unsigned* dst, int nelem) {
    int n2 = nelem >> 1;
    cvt_bf16_kernel<<<dim3((n2 + 255) / 256), blk, 0, stream>>>((const float2*)src, dst, n2);
  };
  auto gemm = [&](const unsigned* a2, const unsigned* w2p, const float* bias, float* c,
                  int N, int K, int relu) {
    gemm_bf16_kernel<<<dim3(M / GTM, N / GTN), blk, 0, stream>>>(a2, w2p, bias, c, M, N, K, relu);
  };

  // Q, K, V projections (x converted once, reused 3x)
  cvt(x, abuf, M * H);
  cvt(wq, wbuf, H * H);  gemm(abuf, wbuf, bq, qb, H, H, 0);
  cvt(wk, wbuf, H * H);  gemm(abuf, wbuf, bk, kb, H, H, 0);
  cvt(wv, wbuf, H * H);  gemm(abuf, wbuf, bv, vb, H, H, 0);
  // attention
  attn_kernel<<<dim3(B * NH, S / 64), dim3(128), 0, stream>>>(qb, kb, vb, mask, attn, S, H, NH);
  // output projection
  cvt(attn, abuf, M * H);
  cvt(wo, wbuf, H * H);  gemm(abuf, wbuf, bo, t1, H, H, 0);
  // sublayer 0
  add_ln_kernel<<<dim3(M), blk, 0, stream>>>(x, t1, g1, be1, x1, H);
  // FFN
  cvt(x1, abuf, M * H);
  cvt(w1, wbuf, 2 * H * H);  gemm(abuf, wbuf, b1, ff1, 2 * H, H, 1);
  cvt(ff1, abuf, M * 2 * H);
  cvt(w2, wbuf, 2 * H * H);  gemm(abuf, wbuf, b2, t1, H, 2 * H, 0);
  // sublayer 1
  add_ln_kernel<<<dim3(M), blk, 0, stream>>>(x1, t1, g2, be2, out, H);
}
